// CGIteration_26414048870871
// MI455X (gfx1250) — matrix operand order repaired and checked by reading, not compiled, not run on hardware
//
#include <hip/hip_runtime.h>

// ---------------------------------------------------------------------------
// CG tensor product + channel mixing for MI455X (gfx1250, wave32, WMMA).
//
//   out_l = f1_l + tp_l @ (U_l @ W_l)        (GEMM-associativity fusion)
//
// Big GEMMs run as v_wmma_f32_16x16x32_bf16 (fp32 accumulate). Tensor-product
// A-tiles are built in registers in the 16-bit WMMA A lane layout, shared
// across the workgroup's 8 waves through LDS.
//
// Workspace layout (requires ws_size >= ~38.1 MB):
//   [0)            f1 in bf16 (all l, same layout as inputs)   16,777,216 B
//   [16777216)     f2 in bf16                                  16,777,216 B
//   [33554432)     UW^T per l: bf16 [256 x cin_l]               4,456,448 B
//   [38010880)     CG tables: 34 paths x 343 floats                46,648 B
// ---------------------------------------------------------------------------

typedef __bf16 bf16;
typedef __attribute__((ext_vector_type(8)))  __bf16 v8bf;
typedef __attribute__((ext_vector_type(16))) __bf16 v16bf;
typedef __attribute__((ext_vector_type(8)))  float  v8f;

#define PATH_L1_INIT {0,1,2,3, 0,1,1,1,2,2,2,3,3, 0,1,1,1,2,2,2,2,3,3,3, 0,1,1,2,2,2,3,3,3,3}
#define PATH_L2_INIT {0,1,2,3, 1,0,1,2,1,2,3,2,3, 2,1,2,3,0,1,2,3,1,2,3, 3,2,3,1,2,3,0,1,2,3}
#define PATH_LO_INIT {0,0,0,0, 1,1,1,1,1,1,1,1,1, 2,2,2,2,2,2,2,2,2,2,2, 3,3,3,3,3,3,3,3,3,3}

constexpr int PATH_L1C[34]  = PATH_L1_INIT;
constexpr int PATH_L2C[34]  = PATH_L2_INIT;
constexpr int PATH_BASEC[5] = {0, 4, 13, 24, 34};
constexpr int NPATHC[4]     = {4, 9, 11, 10};
constexpr int CINC[4]       = {1024, 2304, 2816, 2560};
// element offsets of each l-section inside f1/f2/out (N * (2l+1) * 256)
constexpr long F_OFFC[4]    = {0, 524288, 2097152, 4718592};
// element offsets of each l-section inside UW^T
constexpr long UWT_OFFC[4]  = {0, 262144, 851968, 1572864};

constexpr size_t WS_F1B = 0;
constexpr size_t WS_F2B = 16777216;
constexpr size_t WS_UWT = 33554432;
constexpr size_t WS_CG  = 38010880;   // 33554432 + 8704*256*2

// runtime-indexable device copies (used by cg_init)
__device__ const int d_PATH_L1[34] = PATH_L1_INIT;
__device__ const int d_PATH_L2[34] = PATH_L2_INIT;
__device__ const int d_PATH_LO[34] = PATH_LO_INIT;

// ---------------------------------------------------------------------------
// Pass 0a: Clebsch-Gordan tables via the Racah formula (tiny, runs once).
// Layout: cg[p*343 + m1*49 + m2*7 + m]  (fixed 7x7x7 stride per path).
// ---------------------------------------------------------------------------
__device__ double dfactorial(int n) {
    double r = 1.0;
    for (int i = 2; i <= n; ++i) r *= (double)i;
    return r;
}

__device__ float cg_coeff(int l1, int m1, int l2, int m2, int l3, int m3) {
    if (m3 != m1 + m2) return 0.f;
    double pref = (2.0 * l3 + 1.0) * dfactorial(l3 + l1 - l2) * dfactorial(l3 - l1 + l2)
                * dfactorial(l1 + l2 - l3) / dfactorial(l1 + l2 + l3 + 1);
    pref = sqrt(pref);
    pref *= sqrt(dfactorial(l3 + m3) * dfactorial(l3 - m3) * dfactorial(l1 - m1)
               * dfactorial(l1 + m1) * dfactorial(l2 - m2) * dfactorial(l2 + m2));
    double s = 0.0;
    for (int k = 0; k <= l1 + l2 - l3; ++k) {
        int d0 = k, d1 = l1 + l2 - l3 - k, d2 = l1 - m1 - k, d3 = l2 + m2 - k;
        int d4 = l3 - l2 + m1 + k, d5 = l3 - l1 - m2 + k;
        if (d0 < 0 || d1 < 0 || d2 < 0 || d3 < 0 || d4 < 0 || d5 < 0) continue;
        double den = dfactorial(d0) * dfactorial(d1) * dfactorial(d2)
                   * dfactorial(d3) * dfactorial(d4) * dfactorial(d5);
        s += ((k & 1) ? -1.0 : 1.0) / den;
    }
    return (float)(pref * s);
}

__global__ void cg_init(float* __restrict__ cg) {
    int t = blockIdx.x * 256 + threadIdx.x;
    if (t >= 34 * 343) return;
    int p = t / 343, r = t - p * 343;
    int m1 = r / 49; int rem = r - m1 * 49;
    int m2 = rem / 7; int m = rem - m2 * 7;
    int l1 = d_PATH_L1[p], l2 = d_PATH_L2[p], l = d_PATH_LO[p];
    float v = 0.f;
    if (m1 <= 2 * l1 && m2 <= 2 * l2 && m <= 2 * l)
        v = cg_coeff(l1, m1 - l1, l2, m2 - l2, l, m - l);
    cg[t] = v;
}

// ---------------------------------------------------------------------------
// Pass 0b: fp32 -> bf16 conversion (roofline: bf16 WMMA is the fast path).
// ---------------------------------------------------------------------------
__global__ void f32_to_bf16(const float* __restrict__ src, bf16* __restrict__ dst, int n) {
    int i = blockIdx.x * 256 + threadIdx.x;
    if (i < n) dst[i] = (bf16)src[i];
}

// ---------------------------------------------------------------------------
// Pass 0c: UW^T[l][ko][c] = sum_j U[c][j] * W[j][ko], stored bf16.
// One block per c: U-row reads are wave-uniform (scalar), W reads coalesced.
// ---------------------------------------------------------------------------
__global__ void uw_kernel(const float* __restrict__ U, const float* __restrict__ W,
                          bf16* __restrict__ uwt, int cin) {
    int c = blockIdx.x;
    int ko = threadIdx.x;
    float s = 0.f;
    for (int j = 0; j < 256; ++j)
        s = fmaf(U[(size_t)c * 256 + j], W[(size_t)j * 256 + ko], s);
    uwt[(size_t)ko * cin + c] = (bf16)s;
}

// ---------------------------------------------------------------------------
// Main fused kernel.
// ---------------------------------------------------------------------------
template<int L>
struct Ctx {
    const bf16* __restrict__ f1b;
    const bf16* __restrict__ f2b;
    const bf16* __restrict__ uwrow;   // UW^T row for this wave's kout column
    const float* __restrict__ cg;
    bf16* Ash;
    int n;      // n-row this lane owns in the A fragment (blockIdx.x*16 + (lane&15))
    int lane;
    int lh;     // lane >> 4
    int w;      // wave id (0..7)
};

__device__ __forceinline__ v16bf cat16(v8bf a, v8bf b) {
    return __builtin_shufflevector(a, b, 0, 1, 2, 3, 4, 5, 6, 7,
                                         8, 9, 10, 11, 12, 13, 14, 15);
}

template<int L, int P>
__device__ __forceinline__ void path_body(const Ctx<L>& c, v8f (&acc)[2 * L + 1]) {
    constexpr int l1 = PATH_L1C[P], l2 = PATH_L2C[P];
    constexpr int M1 = 2 * l1 + 1, M2 = 2 * l2 + 1, MO = 2 * L + 1;
    constexpr int PI = P - PATH_BASEC[L];

    const bf16* f1p = c.f1b + F_OFFC[l1] + (size_t)c.n * (M1 * 256);
    const bf16* f2p = c.f2b + F_OFFC[l2] + (size_t)c.n * (M2 * 256);

    for (int kc = 0; kc < 256; kc += 32) {
        // ---- Phase 1: wave m builds the bf16 A-tile (tp fragment) for output m.
        // 16-bit A layout: lane&15 = row; k = kc + (e&7) + 2*(e&8) + 8*(lane>>4).
#pragma unroll
        for (int m = 0; m < MO; ++m) {
            if (c.w != m) continue;
            float a[16];
#pragma unroll
            for (int e = 0; e < 16; ++e) a[e] = 0.f;
#pragma unroll
            for (int m1 = 0; m1 < M1; ++m1) {
                const int mm2 = (m - L) - (m1 - l1);          // compile-time folded
                if (mm2 < -l2 || mm2 > l2) continue;
                const int m2 = mm2 + l2;
                float cgv = c.cg[P * 343 + m1 * 49 + m2 * 7 + m];  // uniform -> s_load
                const bf16* p1 = f1p + m1 * 256 + kc + 8 * c.lh;
                const bf16* p2 = f2p + m2 * 256 + kc + 8 * c.lh;
                v8bf x1l = *(const v8bf*)p1, x1h = *(const v8bf*)(p1 + 16);
                v8bf x2l = *(const v8bf*)p2, x2h = *(const v8bf*)(p2 + 16);
#pragma unroll
                for (int e = 0; e < 8; ++e) {
                    a[e]     = fmaf(cgv, (float)x1l[e] * (float)x2l[e], a[e]);
                    a[e + 8] = fmaf(cgv, (float)x1h[e] * (float)x2h[e], a[e + 8]);
                }
            }
            v8bf o0, o1;
#pragma unroll
            for (int e = 0; e < 8; ++e) { o0[e] = (bf16)a[e]; o1[e] = (bf16)a[e + 8]; }
            bf16* dst = c.Ash + ((size_t)m * 32 + c.lane) * 16;
            *(v8bf*)dst       = o0;
            *(v8bf*)(dst + 8) = o1;
        }
        __syncthreads();

        // ---- Phase 2: every wave: its own B fragment x all shared A fragments.
        // B 32x16 bf16 layout: N = lane&15 (UW^T row), K = (e&7) + 2*(e&8) + 8*(lane>>4).
        const bf16* bp = c.uwrow + PI * 256 + kc + 8 * c.lh;
        v16bf B = cat16(*(const v8bf*)bp, *(const v8bf*)(bp + 16));
#pragma unroll
        for (int m = 0; m < MO; ++m) {
            const bf16* ap = c.Ash + ((size_t)m * 32 + c.lane) * 16;
            v16bf A = cat16(*(const v8bf*)ap, *(const v8bf*)(ap + 8));
            acc[m] = __builtin_amdgcn_wmma_f32_16x16x32_bf16(
                false, A, false, B, (short)0, acc[m], false, false);
        }
        __syncthreads();
    }
}

template<int L, int PI>
__device__ __forceinline__ void path_rec(const Ctx<L>& c, v8f (&acc)[2 * L + 1]) {
    if constexpr (PI < NPATHC[L]) {
        path_body<L, PATH_BASEC[L] + PI>(c, acc);
        path_rec<L, PI + 1>(c, acc);
    }
}

template<int L>
__global__ __launch_bounds__(256) void cg_main(
    const bf16* __restrict__ f1b, const bf16* __restrict__ f2b,
    const bf16* __restrict__ uwt, const float* __restrict__ cg,
    const float* __restrict__ f1f, float* __restrict__ out) {
    constexpr int MO = 2 * L + 1;
    __shared__ bf16 Ash[MO * 32 * 16];

    const int lane = threadIdx.x & 31;
    const int w    = threadIdx.x >> 5;      // 8 waves per workgroup
    const int lr   = lane & 15;
    const int lh   = lane >> 4;
    const int ko   = blockIdx.y * 128 + w * 16 + lr;   // this wave's kout column

    Ctx<L> c;
    c.f1b = f1b; c.f2b = f2b;
    c.uwrow = uwt + UWT_OFFC[L] + (size_t)ko * CINC[L];
    c.cg = cg; c.Ash = Ash;
    c.n = blockIdx.x * 16 + lr;
    c.lane = lane; c.lh = lh; c.w = w;

    v8f acc[MO];
#pragma unroll
    for (int m = 0; m < MO; ++m) { v8f z = {}; acc[m] = z; }

    path_rec<L, 0>(c, acc);

    // Epilogue: residual add. f32 C/D layout: N = lane&15, M = v + 8*(lane>>4).
#pragma unroll
    for (int m = 0; m < MO; ++m) {
#pragma unroll
        for (int v = 0; v < 8; ++v) {
            const int nr = blockIdx.x * 16 + v + 8 * lh;
            const size_t idx = ((size_t)nr * MO + m) * 256 + ko;
            out[idx] = f1f[idx] + acc[m][v];
        }
    }
}

// ---------------------------------------------------------------------------
// Host launcher.
// ---------------------------------------------------------------------------
extern "C" void kernel_launch(void* const* d_in, const int* in_sizes, int n_in,
                              void* d_out, int out_size, void* d_ws, size_t ws_size,
                              hipStream_t stream) {
    (void)in_sizes; (void)n_in; (void)out_size; (void)ws_size;

    const float* f1[4]; const float* f2[4]; const float* U[4]; const float* W[4];
    for (int l = 0; l < 4; ++l) {
        f1[l] = (const float*)d_in[l];
        f2[l] = (const float*)d_in[4 + l];
        U[l]  = (const float*)d_in[8 + l];
        W[l]  = (const float*)d_in[12 + l];
    }
    char* ws   = (char*)d_ws;
    bf16* f1b  = (bf16*)(ws + WS_F1B);
    bf16* f2b  = (bf16*)(ws + WS_F2B);
    bf16* uwt  = (bf16*)(ws + WS_UWT);
    float* cg  = (float*)(ws + WS_CG);
    float* out = (float*)d_out;

    cg_init<<<(34 * 343 + 255) / 256, 256, 0, stream>>>(cg);
    for (int l = 0; l < 4; ++l) {
        int cnt = 2048 * (2 * l + 1) * 256;
        f32_to_bf16<<<(cnt + 255) / 256, 256, 0, stream>>>(f1[l], f1b + F_OFFC[l], cnt);
        f32_to_bf16<<<(cnt + 255) / 256, 256, 0, stream>>>(f2[l], f2b + F_OFFC[l], cnt);
        uw_kernel<<<CINC[l], 256, 0, stream>>>(U[l], W[l], uwt + UWT_OFFC[l], CINC[l]);
    }

    dim3 grid(128, 2), block(256);
    cg_main<0><<<grid, block, 0, stream>>>(f1b, f2b, uwt, cg, f1[0], out + F_OFFC[0]);
    cg_main<1><<<grid, block, 0, stream>>>(f1b, f2b, uwt, cg, f1[1], out + F_OFFC[1]);
    cg_main<2><<<grid, block, 0, stream>>>(f1b, f2b, uwt, cg, f1[2], out + F_OFFC[2]);
    cg_main<3><<<grid, block, 0, stream>>>(f1b, f2b, uwt, cg, f1[3], out + F_OFFC[3]);
}
